// RelationModule_78812649882396
// MI455X (gfx1250) — matrix operand order repaired and checked by reading, not compiled, not run on hardware
//
#include <hip/hip_runtime.h>
#include <hip/hip_bf16.h>
#include <math.h>

// Problem constants (from reference): BS=64, N=256, T=32, D=512, R=8, DK=64
#define BSZ   64
#define NOBJ  256
#define TLEN  32
#define DIMF  512
#define NREL  8
#define DKF   64
#define NN    (NOBJ * NOBJ)   // 65536

// Padded pitches (halves) for transposed B tiles in LDS: multiples of 8
// halves (16B) keep each lane's 16-half fragment read b128-vectorizable;
// 40/264 also spread LDS banks (20/132-dword strides).
#define BPITCH   40    // for 32-deep K tiles (k_proj, k_tmap)
#define VPITCH   264   // for the 256-deep wv^T tile (k_attn)

typedef __attribute__((ext_vector_type(16))) _Float16 v16h;
typedef __attribute__((ext_vector_type(8)))  _Float16 v8h;
typedef __attribute__((ext_vector_type(8)))  float    v8f;
typedef __attribute__((ext_vector_type(4)))  int      v4i;

#if defined(__gfx1250__) && __has_builtin(__builtin_amdgcn_global_load_async_to_lds_b128) && __has_builtin(__builtin_amdgcn_s_wait_asynccnt)
#define USE_ASYNC_LDS 1
#else
#define USE_ASYNC_LDS 0
#endif

// ---------------------------------------------------------------------------
// WMMA helper: D(16x16,f32) = A(16x32,f16) x B(32x16,f16) + C
// ---------------------------------------------------------------------------
__device__ __forceinline__ v8f wmma_f16(v16h a, v16h b, v8f c) {
  return __builtin_amdgcn_wmma_f32_16x16x32_f16(
      /*neg_a=*/false, a, /*neg_b=*/false, b,
      /*c_mod=*/(short)0, c, /*reuse_a=*/false, /*reuse_b=*/false);
}

// A fragment (16x32 f16, row-major source, leading dim ld):
// lane holds row m = lane&15; halves i<8 -> k = kb+i, i>=8 -> k = 16+kb+(i-8),
// kb = (lane<16 ? 0 : 8)   [ISA 7.12.2, 16-bit A 16x32 table]
__device__ __forceinline__ v16h load_frag_a(const _Float16* base, int ld) {
  const int lane = threadIdx.x & 31;
  const _Float16* p = base + (lane & 15) * ld;
  const int kb = (lane < 16) ? 0 : 8;
  v16h a;
#pragma unroll
  for (int i = 0; i < 8; ++i) a[i] = p[kb + i];
#pragma unroll
  for (int i = 0; i < 8; ++i) a[8 + i] = p[16 + kb + i];
  return a;
}

// Same A fragment but f32 source (converted on the fly) — attn probabilities.
__device__ __forceinline__ v16h load_frag_a_f32(const float* base, int ld) {
  const int lane = threadIdx.x & 31;
  const float* p = base + (lane & 15) * ld;
  const int kb = (lane < 16) ? 0 : 8;
  v16h a;
#pragma unroll
  for (int i = 0; i < 8; ++i) a[i] = (_Float16)p[kb + i];
#pragma unroll
  for (int i = 0; i < 8; ++i) a[8 + i] = (_Float16)p[16 + kb + i];
  return a;
}

// B fragment (32x16) from an N-major ("transposed") source: logical
// B[k][n] = src[n*ld + k]. Lane holds column n = lane&15; halves i<8 -> k = kb+i,
// i>=8 -> k = 16+kb+(i-8), kb = (lane<16 ? 0 : 8)  -> 2x ds_load_b128.
__device__ __forceinline__ v16h load_frag_bt(const _Float16* base, int ld) {
  const int lane = threadIdx.x & 31;
  const _Float16* p = base + (lane & 15) * ld + ((lane < 16) ? 0 : 8);
  v16h b;
#pragma unroll
  for (int i = 0; i < 8; ++i) b[i] = p[i];
#pragma unroll
  for (int i = 0; i < 8; ++i) b[8 + i] = p[16 + i];
  return b;
}

#if USE_ASYNC_LDS
// 16-byte async copy HBM -> LDS (gfx1250 GLOBAL_LOAD_ASYNC_TO_LDS_B128).
// Builtin signature (from hipcc diagnostic): (v4i AS1*, v4i AS3*, imm, imm).
__device__ __forceinline__ void async_copy16(const void* g, void* l) {
  __builtin_amdgcn_global_load_async_to_lds_b128(
      (__attribute__((address_space(1))) v4i*)(uintptr_t)g,
      (__attribute__((address_space(3))) v4i*)(unsigned)(uintptr_t)l,
      0, 0);
}
#endif

// ---------------------------------------------------------------------------
// Kernel 1: text summary branch. grid=BS, block=256.
// ---------------------------------------------------------------------------
__global__ void __launch_bounds__(256)
k_text(const float* __restrict__ t, const float* __restrict__ t_mask,
       const float* __restrict__ tc_w, const float* __restrict__ tc_b,
       float* __restrict__ tsum) {
  const int bs = blockIdx.x;
  const int tid = threadIdx.x;
  __shared__ float score[TLEN];
  if (tid < TLEN) {
    const float m = t_mask[bs * TLEN + tid];
    const float* row = t + (size_t)(bs * TLEN + tid) * DIMF;
    float acc = 0.f;
    for (int d = 0; d < DIMF; ++d) acc += row[d] * tc_w[d];
    acc = acc * m + tc_b[0];
    score[tid] = (m == 0.f) ? 0.f : 1.f / (1.f + __expf(-acc));
  }
  __syncthreads();
  for (int d = tid; d < DIMF; d += 256) {
    float s = 0.f;
#pragma unroll 4
    for (int j = 0; j < TLEN; ++j) {
      const float m = t_mask[bs * TLEN + j];
      s += t[(size_t)(bs * TLEN + j) * DIMF + d] * m * score[j];
    }
    tsum[bs * DIMF + d] = s;
  }
}

// ---------------------------------------------------------------------------
// Kernel 2: tmap = relu(tsum(64x512) @ tmap_w(512x65536) + tmap_b).
// ---------------------------------------------------------------------------
__global__ void __launch_bounds__(256)
k_tmap(const float* __restrict__ tsum, const float* __restrict__ Wt,
       const float* __restrict__ bt, float* __restrict__ tmap) {
  extern __shared__ char smem[];
  _Float16* sA  = (_Float16*)smem;                      // 64 x 512    (64 KB)
  _Float16* sBt = (_Float16*)(smem + BSZ * DIMF * 2);   // 128 x BPITCH
  const int tid = threadIdx.x, wave = tid >> 5, lane = tid & 31;
  const int nBase = blockIdx.x * 128;

  for (int ch = tid; ch < (BSZ * DIMF) / 4; ch += 256) {
    const float4 a4 = *(const float4*)&tsum[ch * 4];
    _Float16* d = sA + ch * 4;
    d[0] = (_Float16)a4.x; d[1] = (_Float16)a4.y;
    d[2] = (_Float16)a4.z; d[3] = (_Float16)a4.w;
  }
  __syncthreads();

  v8f acc[4] = {v8f{}, v8f{}, v8f{}, v8f{}};
  for (int ks = 0; ks < DIMF; ks += 32) {
    for (int ch = tid; ch < (32 * 128) / 4; ch += 256) {
      const int rr = ch >> 5;             // k row 0..31
      const int cc = (ch & 31) * 4;       // col  0..124
      const float4 w4 = *(const float4*)&Wt[(size_t)(ks + rr) * NN + nBase + cc];
      sBt[(cc + 0) * BPITCH + rr] = (_Float16)w4.x;
      sBt[(cc + 1) * BPITCH + rr] = (_Float16)w4.y;
      sBt[(cc + 2) * BPITCH + rr] = (_Float16)w4.z;
      sBt[(cc + 3) * BPITCH + rr] = (_Float16)w4.w;
    }
    if (ks + 32 < DIMF) {
      const int rr = tid >> 7, cc = tid & 127;
      __builtin_prefetch(&Wt[(size_t)(ks + 32 + rr) * NN + nBase + cc], 0, 1);
    }
    __syncthreads();
    const v16h bf = load_frag_bt(sBt + (wave * 16) * BPITCH, BPITCH);
#pragma unroll
    for (int mt = 0; mt < 4; ++mt) {
      const v16h af = load_frag_a(sA + (mt * 16) * DIMF + ks, DIMF);
      acc[mt] = wmma_f16(af, bf, acc[mt]);
    }
    __syncthreads();
  }
  const int rowOff = (lane < 16) ? 0 : 8;
#pragma unroll
  for (int mt = 0; mt < 4; ++mt)
#pragma unroll
    for (int j = 0; j < 8; ++j) {
      const int row = mt * 16 + j + rowOff;
      const int col = nBase + wave * 16 + (lane & 15);
      tmap[(size_t)row * NN + col] = fmaxf(acc[mt][j] + bt[col], 0.f);
    }
}

// ---------------------------------------------------------------------------
// Kernel 3: projection  Y(f16) = (X(16384x512) @ W(512x512) + bias)[*rowmask]
// ---------------------------------------------------------------------------
__global__ void __launch_bounds__(256)
k_proj(const float* __restrict__ X, const float* __restrict__ W,
       const float* __restrict__ bias, const float* __restrict__ rowmask,
       _Float16* __restrict__ Y) {
  __shared__ _Float16 sA[128 * 32];
  __shared__ _Float16 sBt[64 * BPITCH];
  const int tid = threadIdx.x, wave = tid >> 5, lane = tid & 31;
  const int mBase = blockIdx.x * 128;
  const int nBase = blockIdx.y * 64;
  const int mg = wave & 3;
  const int ng = wave >> 2;
  v8f acc[2][2] = {{v8f{}, v8f{}}, {v8f{}, v8f{}}};

  for (int ks = 0; ks < DIMF; ks += 32) {
    for (int ch = tid; ch < (128 * 32) / 4; ch += 256) {
      const int rr = ch >> 3, cc = (ch & 7) * 4;
      const float4 x4 = *(const float4*)&X[(size_t)(mBase + rr) * DIMF + ks + cc];
      _Float16* d = sA + rr * 32 + cc;
      d[0] = (_Float16)x4.x; d[1] = (_Float16)x4.y;
      d[2] = (_Float16)x4.z; d[3] = (_Float16)x4.w;
    }
    for (int ch = tid; ch < (32 * 64) / 4; ch += 256) {
      const int rr = ch >> 4, cc = (ch & 15) * 4;    // rr = k, cc = n
      const float4 w4 = *(const float4*)&W[(size_t)(ks + rr) * DIMF + nBase + cc];
      sBt[(cc + 0) * BPITCH + rr] = (_Float16)w4.x;
      sBt[(cc + 1) * BPITCH + rr] = (_Float16)w4.y;
      sBt[(cc + 2) * BPITCH + rr] = (_Float16)w4.z;
      sBt[(cc + 3) * BPITCH + rr] = (_Float16)w4.w;
    }
    __syncthreads();
    const v16h a0 = load_frag_a(sA + (mg * 32 + 0) * 32, 32);
    const v16h a1 = load_frag_a(sA + (mg * 32 + 16) * 32, 32);
    const v16h b0 = load_frag_bt(sBt + (ng * 32 + 0) * BPITCH, BPITCH);
    const v16h b1 = load_frag_bt(sBt + (ng * 32 + 16) * BPITCH, BPITCH);
    acc[0][0] = wmma_f16(a0, b0, acc[0][0]);
    acc[0][1] = wmma_f16(a0, b1, acc[0][1]);
    acc[1][0] = wmma_f16(a1, b0, acc[1][0]);
    acc[1][1] = wmma_f16(a1, b1, acc[1][1]);
    __syncthreads();
  }
  const int rowOff = (lane < 16) ? 0 : 8;
#pragma unroll
  for (int i = 0; i < 2; ++i)
#pragma unroll
    for (int jj = 0; jj < 2; ++jj)
#pragma unroll
      for (int j = 0; j < 8; ++j) {
        const int row = mBase + mg * 32 + i * 16 + j + rowOff;
        const int col = nBase + ng * 32 + jj * 16 + (lane & 15);
        float o = acc[i][jj][j] + bias[col];
        if (rowmask) o *= rowmask[row];
        Y[(size_t)row * DIMF + col] = (_Float16)o;
      }
}

// ---------------------------------------------------------------------------
// Kernel 4: fused attention per (bs, r). grid=(64,8), block=256 (8 waves).
// LDS: wg1/wg2/wk/wq (4 x 256x64 f16, async-copied) + wv^T (64x264 f16)
//      + 8 x (16x256 f32) softmax scratch  = ~289 KB (CDNA5 WGP: 320 KB).
// ---------------------------------------------------------------------------
__global__ void __launch_bounds__(256)
k_attn(const _Float16* __restrict__ wg1, const _Float16* __restrict__ wg2,
       const _Float16* __restrict__ wk,  const _Float16* __restrict__ wq,
       const _Float16* __restrict__ wv,
       const float* __restrict__ tmap, const float* __restrict__ vin,
       const float* __restrict__ vmask, float* __restrict__ out) {
  const int bs = blockIdx.x, r = blockIdx.y;
  extern __shared__ char smem[];
  _Float16* mWg1 = (_Float16*)smem;
  _Float16* mWg2 = mWg1 + NOBJ * DKF;
  _Float16* mWk  = mWg2 + NOBJ * DKF;
  _Float16* mWq  = mWk  + NOBJ * DKF;
  _Float16* mWvT = mWq  + NOBJ * DKF;                   // 64 x VPITCH
  float* scratch = (float*)(mWvT + DKF * VPITCH);       // 8 * 16 * 256 f32
  const int tid = threadIdx.x, wave = tid >> 5, lane = tid & 31;
  const int rowOff = (lane < 16) ? 0 : 8;

  // ---- stage operands (relation-r 64-col slices) ----
  const size_t base = (size_t)(bs * NOBJ) * DIMF + (size_t)r * DKF;
#if USE_ASYNC_LDS
  for (int ch = tid; ch < (NOBJ * DKF) / 8; ch += 256) {   // 2048 x 16B
    const int n = ch >> 3, kc = (ch & 7) * 8;
    const size_t src = base + (size_t)n * DIMF + kc;
    async_copy16(wg1 + src, mWg1 + ch * 8);
    async_copy16(wg2 + src, mWg2 + ch * 8);
    async_copy16(wk  + src, mWk  + ch * 8);
    async_copy16(wq  + src, mWq  + ch * 8);
  }
#else
  for (int ch = tid; ch < (NOBJ * DKF) / 8; ch += 256) {
    const int n = ch >> 3, kc = (ch & 7) * 8;
    const size_t src = base + (size_t)n * DIMF + kc;
    *(v8h*)(mWg1 + ch * 8) = *(const v8h*)(wg1 + src);
    *(v8h*)(mWg2 + ch * 8) = *(const v8h*)(wg2 + src);
    *(v8h*)(mWk  + ch * 8) = *(const v8h*)(wk  + src);
    *(v8h*)(mWq  + ch * 8) = *(const v8h*)(wq  + src);
  }
#endif
  // wv staged transposed: mWvT[c][m] = wv[m][c]
  for (int ch = tid; ch < (NOBJ * DKF) / 8; ch += 256) {
    const int n = ch >> 3, kc = (ch & 7) * 8;
    const v8h w8 = *(const v8h*)(wv + base + (size_t)n * DIMF + kc);
#pragma unroll
    for (int q = 0; q < 8; ++q) mWvT[(kc + q) * VPITCH + n] = w8[q];
  }
#if USE_ASYNC_LDS
  __builtin_amdgcn_s_wait_asynccnt(0);
#endif
  __syncthreads();

  float* scr = scratch + wave * (16 * NOBJ);
  const float inv_sqrt_dk = 0.125f;   // 1/sqrt(64)

  for (int pass = 0; pass < 2; ++pass) {
    const int n0 = (pass * 8 + wave) * 16;

    // ---- phase A: fused score tiles -> scratch logits ----
    const v16h ak0 = load_frag_a(mWk  + n0 * DKF + 0,  DKF);
    const v16h ak1 = load_frag_a(mWk  + n0 * DKF + 32, DKF);
    const v16h ag0 = load_frag_a(mWg1 + n0 * DKF + 0,  DKF);
    const v16h ag1 = load_frag_a(mWg1 + n0 * DKF + 32, DKF);
    for (int mt = 0; mt < 16; ++mt) {
      v8f accd = v8f{};
      v8f accg = v8f{};
      const v16h bq0 = load_frag_bt(mWq  + (mt * 16) * DKF + 0,  DKF);
      const v16h bq1 = load_frag_bt(mWq  + (mt * 16) * DKF + 32, DKF);
      const v16h bg0 = load_frag_bt(mWg2 + (mt * 16) * DKF + 0,  DKF);
      const v16h bg1 = load_frag_bt(mWg2 + (mt * 16) * DKF + 32, DKF);
      accd = wmma_f16(ak0, bq0, accd);
      accd = wmma_f16(ak1, bq1, accd);
      accg = wmma_f16(ag0, bg0, accg);
      accg = wmma_f16(ag1, bg1, accg);
#pragma unroll
      for (int j = 0; j < 8; ++j) {
        const int lr  = j + rowOff;
        const int row = n0 + lr;
        const int col = mt * 16 + (lane & 15);
        float g = fmaxf(accg[j], 0.f) + tmap[(size_t)bs * NN + row * NOBJ + col];
        float val = __logf(fmaxf(g, 1e-6f)) + accd[j] * inv_sqrt_dk;
        if (vmask[bs * NOBJ + col] == 0.f) val = -__builtin_inff();
        scr[lr * NOBJ + col] = val;
      }
    }
    __syncthreads();

    // ---- phase B: row softmax (one row per lane<16) ----
    if (lane < 16) {
      float* rowp = scr + lane * NOBJ;
      float mx = -__builtin_inff();
      for (int m = 0; m < NOBJ; ++m) mx = fmaxf(mx, rowp[m]);
      float sum = 0.f;
      for (int m = 0; m < NOBJ; ++m) {
        const float e = __expf(rowp[m] - mx);
        rowp[m] = e;
        sum += e;
      }
      const float inv = 1.f / sum;
      for (int m = 0; m < NOBJ; ++m) rowp[m] *= inv;
    }
    __syncthreads();

    // ---- phase C: out strip = attn @ wv, plus residual ----
    v8f acco[4] = {v8f{}, v8f{}, v8f{}, v8f{}};
    for (int ms = 0; ms < NOBJ; ms += 32) {
      const v16h aat = load_frag_a_f32(scr + ms, NOBJ);
#pragma unroll
      for (int ct = 0; ct < 4; ++ct) {
        const v16h bv = load_frag_bt(mWvT + (ct * 16) * VPITCH + ms, VPITCH);
        acco[ct] = wmma_f16(aat, bv, acco[ct]);
      }
    }
#pragma unroll
    for (int ct = 0; ct < 4; ++ct)
#pragma unroll
      for (int j = 0; j < 8; ++j) {
        const int row = n0 + j + rowOff;
        const int col = ct * 16 + (lane & 15);
        const size_t di = ((size_t)(bs * NOBJ + row)) * DIMF + r * DKF + col;
        out[di] = acco[ct][j] + vin[di];
      }
    __syncthreads();
  }
}

// ---------------------------------------------------------------------------
// Host-side launch
// ---------------------------------------------------------------------------
extern "C" void kernel_launch(void* const* d_in, const int* in_sizes, int n_in,
                              void* d_out, int out_size, void* d_ws, size_t ws_size,
                              hipStream_t stream) {
  (void)in_sizes; (void)n_in; (void)out_size; (void)ws_size;
  const float* v      = (const float*)d_in[0];
  const float* b      = (const float*)d_in[1];
  const float* v_mask = (const float*)d_in[2];
  const float* t      = (const float*)d_in[3];
  const float* t_mask = (const float*)d_in[4];
  const float* WG1_w  = (const float*)d_in[5];
  const float* WG1_b  = (const float*)d_in[6];
  const float* WG2_w  = (const float*)d_in[7];
  const float* WG2_b  = (const float*)d_in[8];
  const float* WK_w   = (const float*)d_in[9];
  const float* WK_b   = (const float*)d_in[10];
  const float* WQ_w   = (const float*)d_in[11];
  const float* WQ_b   = (const float*)d_in[12];
  const float* WV_w   = (const float*)d_in[13];
  const float* WV_b   = (const float*)d_in[14];
  const float* tc_w   = (const float*)d_in[15];
  const float* tc_b   = (const float*)d_in[16];
  const float* tmap_w = (const float*)d_in[17];
  const float* tmap_b = (const float*)d_in[18];
  float* out = (float*)d_out;

  float* tsum    = (float*)d_ws;                         // 64*512 f32
  float* tmapBuf = tsum + BSZ * DIMF;                    // 64*65536 f32
  _Float16* projBase = (_Float16*)(tmapBuf + (size_t)BSZ * NN);
  const size_t projElems = (size_t)BSZ * NOBJ * DIMF;    // 8388608
  _Float16* wg1h = projBase + 0 * projElems;
  _Float16* wg2h = projBase + 1 * projElems;
  _Float16* wkh  = projBase + 2 * projElems;
  _Float16* wqh  = projBase + 3 * projElems;
  _Float16* wvh  = projBase + 4 * projElems;

  const int tmapSmem = BSZ * DIMF * 2 + 128 * BPITCH * 2;               // 75776
  const int attnSmem = 4 * NOBJ * DKF * 2 + DKF * VPITCH * 2
                     + 8 * 16 * NOBJ * 4;                               // 295936
  (void)hipFuncSetAttribute((const void*)k_tmap,
      hipFuncAttributeMaxDynamicSharedMemorySize, tmapSmem);
  (void)hipFuncSetAttribute((const void*)k_attn,
      hipFuncAttributeMaxDynamicSharedMemorySize, attnSmem);

  k_text<<<BSZ, 256, 0, stream>>>(t, t_mask, tc_w, tc_b, tsum);
  k_tmap<<<NN / 128, 256, tmapSmem, stream>>>(tsum, tmap_w, tmap_b, tmapBuf);

  dim3 gp(16384 / 128, DIMF / 64);
  k_proj<<<gp, 256, 0, stream>>>(b, WG1_w, WG1_b, v_mask, wg1h);
  k_proj<<<gp, 256, 0, stream>>>(b, WG2_w, WG2_b, v_mask, wg2h);
  k_proj<<<gp, 256, 0, stream>>>(v, WK_w, WK_b, nullptr, wkh);
  k_proj<<<gp, 256, 0, stream>>>(v, WQ_w, WQ_b, nullptr, wqh);
  k_proj<<<gp, 256, 0, stream>>>(v, WV_w, WV_b, nullptr, wvh);

  k_attn<<<dim3(BSZ, NREL), 256, attnSmem, stream>>>(
      wg1h, wg2h, wkh, wqh, wvh, tmapBuf, v, v_mask, out);
}
